// WaveSTFTGAT_1872605741299
// MI455X (gfx1250) — matrix operand order repaired and checked by reading, not compiled, not run on hardware
//
#include <hip/hip_runtime.h>
#include <hip/hip_bf16.h>
#include <math.h>

// ---------------------------------------------------------------------------
// Types for CDNA5 WMMA bf16 path
// ---------------------------------------------------------------------------
typedef __attribute__((ext_vector_type(16))) __bf16 v16bf;
typedef __attribute__((ext_vector_type(8)))  float  v8f;

#define BM 128
#define BN 128
#define BK 32

// Epilogue modes
#define EP_NONE   0
#define EP_LRELU  1   // leaky-relu slope 0.01 (after bias)
#define EP_CAUSAL 2   // causal mask (j>i -> 0.01*-1e9) then lrelu(0.01)
#define EP_SIGM   3   // sigmoid (after bias)

union Frag { v16bf v; unsigned u[8]; };

// ---------------------------------------------------------------------------
// Staging helpers: two-phase (load-all, then convert+store) so the compiler
// keeps 8 global_load_b128 in flight per thread instead of serializing on
// per-chunk s_wait_loadcnt 0.
// ---------------------------------------------------------------------------
__device__ __forceinline__ void loadA_g(const float* __restrict__ A, int rowBase,
                                        int k0, int lda, int t, float4 pa[4])
{
#pragma unroll
    for (int i = 0; i < 4; ++i) {
        int idx = t + i * 256;              // 1024 float4 chunks of [BM][BK]
        int r   = idx >> 3;
        int k4  = (idx & 7) << 2;
        pa[i] = *(const float4*)(A + (long long)(rowBase + r) * lda + k0 + k4);
    }
}

__device__ __forceinline__ void storeA_s(__bf16* __restrict__ As, int t, const float4 pa[4])
{
#pragma unroll
    for (int i = 0; i < 4; ++i) {
        int idx = t + i * 256;
        int r   = idx >> 3;
        int k4  = (idx & 7) << 2;
        __bf16* dst = &As[r * BK + k4];
        dst[0] = (__bf16)pa[i].x; dst[1] = (__bf16)pa[i].y;
        dst[2] = (__bf16)pa[i].z; dst[3] = (__bf16)pa[i].w;
    }
}

template<int TRANSB, int GUARD>
__device__ __forceinline__ void loadB_g(const float* __restrict__ Bm, int colBase,
                                        int k0, int ldb, int Nc, int t, float4 pb[4])
{
#pragma unroll
    for (int i = 0; i < 4; ++i) {
        int idx = t + i * 256;
        if (TRANSB) {                        // B is [Nc,K]; contiguous along k
            int n   = idx >> 3;
            int k4  = (idx & 7) << 2;
            int col = colBase + n;
            float4 v = {0.f, 0.f, 0.f, 0.f};
            if (!GUARD || col < Nc)
                v = *(const float4*)(Bm + (long long)col * ldb + k0 + k4);
            pb[i] = v;
        } else {                             // B is [K,Nc]; contiguous along n
            int kk  = idx >> 5;
            int n4  = (idx & 31) << 2;
            int col = colBase + n4;
            float4 v;
            if (!GUARD || col + 3 < Nc) {
                v = *(const float4*)(Bm + (long long)(k0 + kk) * ldb + col);
            } else {
                v.x = (col + 0 < Nc) ? Bm[(long long)(k0 + kk) * ldb + col + 0] : 0.f;
                v.y = (col + 1 < Nc) ? Bm[(long long)(k0 + kk) * ldb + col + 1] : 0.f;
                v.z = (col + 2 < Nc) ? Bm[(long long)(k0 + kk) * ldb + col + 2] : 0.f;
                v.w = (col + 3 < Nc) ? Bm[(long long)(k0 + kk) * ldb + col + 3] : 0.f;
            }
            pb[i] = v;
        }
    }
}

template<int TRANSB>
__device__ __forceinline__ void storeB_s(__bf16* __restrict__ Bs, int t, const float4 pb[4])
{
#pragma unroll
    for (int i = 0; i < 4; ++i) {
        int idx = t + i * 256;
        if (TRANSB) {
            int n  = idx >> 3;
            int k4 = (idx & 7) << 2;
            __bf16* dst = &Bs[n * BK + k4];
            dst[0] = (__bf16)pb[i].x; dst[1] = (__bf16)pb[i].y;
            dst[2] = (__bf16)pb[i].z; dst[3] = (__bf16)pb[i].w;
        } else {                             // transpose into Bs[n][k]
            int kk = idx >> 5;
            int n4 = (idx & 31) << 2;
            Bs[(n4 + 0) * BK + kk] = (__bf16)pb[i].x;
            Bs[(n4 + 1) * BK + kk] = (__bf16)pb[i].y;
            Bs[(n4 + 2) * BK + kk] = (__bf16)pb[i].z;
            Bs[(n4 + 3) * BK + kk] = (__bf16)pb[i].w;
        }
    }
}

// ---------------------------------------------------------------------------
// Generic bf16-WMMA GEMM: C = epi(alpha*A*B(^T) [+ C] [+ bias])
// Double-buffered LDS software pipeline, 8 waves, wave tile 32x64 (2x4 WMMA).
// M multiple of 128 and K multiple of 32 for all call sites; Nc guarded only
// in the GUARD=1 instantiation (decoder, Nc=12).
// ---------------------------------------------------------------------------
template<int TRANSB, int GUARD>
__global__ __launch_bounds__(256)
void gemm_bf16_wmma(const float* __restrict__ A, const float* __restrict__ Bm,
                    float* __restrict__ C,
                    int M, int Nc, int K, int lda, int ldb, int ldc,
                    long long sA, long long sB, long long sC,
                    float alpha, int accum,
                    const float* __restrict__ bias, int mode)
{
    __shared__ __bf16 As[2][BM * BK];
    __shared__ __bf16 Bs[2][BN * BK];

    const int bz = blockIdx.z;
    A  += (long long)bz * sA;
    Bm += (long long)bz * sB;
    C  += (long long)bz * sC;

    const int rowBase = blockIdx.y * BM;
    const int colBase = blockIdx.x * BN;

    const int t    = threadIdx.x;
    const int wave = t >> 5;
    const int lane = t & 31;
    const int wrow = wave >> 1;   // 0..3 -> 32-row strip
    const int wcol = wave & 1;    // 0..1 -> 64-col strip
    const int l15  = lane & 15;
    const int hi   = lane >> 4;   // 0 or 1

    v8f acc[2][4];
    const v8f vzero = {0.f,0.f,0.f,0.f,0.f,0.f,0.f,0.f};
#pragma unroll
    for (int i = 0; i < 2; ++i)
#pragma unroll
        for (int j = 0; j < 4; ++j) acc[i][j] = vzero;

    // ---- prologue: stage k0 = 0 into buffer 0 ----
    float4 pa[4], pb[4];
    loadA_g(A, rowBase, 0, lda, t, pa);
    loadB_g<TRANSB, GUARD>(Bm, colBase, 0, ldb, Nc, t, pb);
    storeA_s(As[0], t, pa);
    storeB_s<TRANSB>(Bs[0], t, pb);

    int cur = 0;
    for (int k0 = 0; k0 < K; k0 += BK) {
        const bool more = (k0 + BK) < K;
        if (more) {                 // prefetch next tile while computing this one
            loadA_g(A, rowBase, k0 + BK, lda, t, pa);
            loadB_g<TRANSB, GUARD>(Bm, colBase, k0 + BK, ldb, Nc, t, pb);
        }
        __syncthreads();            // buf[cur] ready; buf[1-cur] free

        const __bf16* Ac = As[cur];
        const __bf16* Bc = Bs[cur];

        // A fragments (16x32 bf16 lane layout)
        Frag afr[2];
        const int kb = hi * 8;
#pragma unroll
        for (int i = 0; i < 2; ++i) {
            const __bf16* base = &Ac[(wrow * 32 + i * 16 + l15) * BK];
#pragma unroll
            for (int q = 0; q < 4; ++q) {
                afr[i].u[q]     = *(const unsigned*)(base + kb + 2 * q);
                afr[i].u[4 + q] = *(const unsigned*)(base + 16 + kb + 2 * q);
            }
        }
        // B fragments + WMMA
#pragma unroll
        for (int j = 0; j < 4; ++j) {
            const __bf16* base = &Bc[(wcol * 64 + j * 16 + l15) * BK] + hi * 8;
            Frag bfr;
#pragma unroll
            for (int q = 0; q < 8; ++q)
                bfr.u[q] = *(const unsigned*)(base + (q >> 2) * 16 + 2 * (q & 3));
#pragma unroll
            for (int i = 0; i < 2; ++i)
                acc[i][j] = __builtin_amdgcn_wmma_f32_16x16x32_bf16(
                    false, afr[i].v, false, bfr.v, (short)0, acc[i][j], false, false);
        }

        if (more) {                 // fill the other buffer
            storeA_s(As[cur ^ 1], t, pa);
            storeB_s<TRANSB>(Bs[cur ^ 1], t, pb);
        }
        cur ^= 1;
    }

    // ---- epilogue ----
#pragma unroll
    for (int i = 0; i < 2; ++i) {
        int tileM = rowBase + wrow * 32 + i * 16;
#pragma unroll
        for (int j = 0; j < 4; ++j) {
            int col = colBase + wcol * 64 + j * 16 + l15;
            if (GUARD && col >= Nc) continue;
            float bv = bias ? bias[col] : 0.0f;
#pragma unroll
            for (int v = 0; v < 8; ++v) {
                int row = tileM + hi * 8 + v;
                long long off = (long long)row * ldc + col;
                float val = alpha * acc[i][j][v];
                if (accum) val += C[off];
                val += bv;
                if (mode == EP_LRELU) {
                    val = val >= 0.f ? val : 0.01f * val;
                } else if (mode == EP_CAUSAL) {
                    if (col > row) val = 0.01f * (-1e9f);
                    else val = val >= 0.f ? val : 0.01f * val;
                } else if (mode == EP_SIGM) {
                    val = 1.0f / (1.0f + expf(-val));
                }
                C[off] = val;
            }
        }
    }
}

// ---------------------------------------------------------------------------
// LayerNorm over last dim (F=512), one wave per row. Modes:
// 0: out = LN(x)        1: out = gelu_exact(LN(x))
// 2: avg = 0.25*y, mx = y        3: avg += 0.25*y, mx = max(mx, y)
// ---------------------------------------------------------------------------
__global__ __launch_bounds__(256)
void ln_kernel(const float* __restrict__ in, const float* __restrict__ g,
               const float* __restrict__ bta, float* __restrict__ out,
               float* __restrict__ avg, float* __restrict__ mx,
               int Fdim, int mode)
{
    const int wave = threadIdx.x >> 5, lane = threadIdx.x & 31;
    const long long row = (long long)blockIdx.x * 8 + wave;
    const float* x = in + row * Fdim;
    float vals[16];
    float s = 0.f;
#pragma unroll
    for (int i = 0; i < 16; ++i) { vals[i] = x[lane + 32 * i]; s += vals[i]; }
#pragma unroll
    for (int o = 16; o >= 1; o >>= 1) s += __shfl_xor(s, o, 32);
    const float mean = s / (float)Fdim;
    float vs = 0.f;
#pragma unroll
    for (int i = 0; i < 16; ++i) { float d = vals[i] - mean; vs += d * d; }
#pragma unroll
    for (int o = 16; o >= 1; o >>= 1) vs += __shfl_xor(vs, o, 32);
    const float rstd = rsqrtf(vs / (float)Fdim + 1e-5f);
#pragma unroll
    for (int i = 0; i < 16; ++i) {
        int c = lane + 32 * i;
        float y = (vals[i] - mean) * rstd * g[c] + bta[c];
        long long off = row * Fdim + c;
        if (mode == 0)      out[off] = y;
        else if (mode == 1) out[off] = 0.5f * y * (1.0f + erff(y * 0.70710678118f));
        else if (mode == 2) { avg[off] = 0.25f * y; mx[off] = y; }
        else                { avg[off] += 0.25f * y; mx[off] = fmaxf(mx[off], y); }
    }
}

// ---------------------------------------------------------------------------
// Dual GEMV: e1[row] = h[row,:].a1, e2[row] = h[row,:].a2, one wave per row
// ---------------------------------------------------------------------------
__global__ __launch_bounds__(256)
void gemv2_kernel(const float* __restrict__ h, const float* __restrict__ a1,
                  const float* __restrict__ a2, float* __restrict__ e1,
                  float* __restrict__ e2, int Fdim)
{
    const int wave = threadIdx.x >> 5, lane = threadIdx.x & 31;
    const long long row = (long long)blockIdx.x * 8 + wave;
    const float* x = h + row * Fdim;
    float s1 = 0.f, s2 = 0.f;
    for (int i = lane; i < Fdim; i += 32) { float v = x[i]; s1 += v * a1[i]; s2 += v * a2[i]; }
#pragma unroll
    for (int o = 16; o >= 1; o >>= 1) { s1 += __shfl_xor(s1, o, 32); s2 += __shfl_xor(s2, o, 32); }
    if (lane == 0) { e1[row] = s1; e2[row] = s2; }
}

// ---------------------------------------------------------------------------
// Elementwise helpers
// ---------------------------------------------------------------------------
__global__ void enc_combine_kernel(const float* __restrict__ ro, const float* __restrict__ io,
                                   const float* __restrict__ ew, const float* __restrict__ eb,
                                   float* __restrict__ x, long long n)
{
    long long i = (long long)blockIdx.x * 256 + threadIdx.x;
    if (i >= n) return;
    float v = ro[i] * ew[0] + io[i] * ew[1] + eb[0];
    x[i] = v >= 0.f ? v : 0.01f * v;
}

__global__ void gat_input_kernel(const float* __restrict__ x, const float* __restrict__ nw,
                                 const float* __restrict__ nb, float* __restrict__ out,
                                 int Nn, int Fdim, long long total)
{
    long long i = (long long)blockIdx.x * 256 + threadIdx.x;
    if (i >= total) return;
    int n = (int)((i / Fdim) % Nn);
    out[i] = x[i] * (1.0f + nw[n]) + nb[n];
}

__global__ void att_build_kernel(const float* __restrict__ e1, const float* __restrict__ e2,
                                 const float* __restrict__ adj, float* __restrict__ att,
                                 int Nn, long long total)
{
    long long i = (long long)blockIdx.x * 256 + threadIdx.x;
    if (i >= total) return;
    int j  = (int)(i % Nn);
    long long bi = i / Nn;
    int ir = (int)(bi % Nn);
    long long b = bi / Nn;
    float e = e1[b * Nn + ir] + e2[b * Nn + j];
    e = e >= 0.f ? e : 0.2f * e;
    float v = adj[(long long)ir * Nn + j] > 0.f ? e : -9e15f;
    att[i] = v >= 0.f ? v : 0.2f * v;
}

__global__ void elu_res_kernel(const float* __restrict__ hp, const float* __restrict__ h,
                               float* __restrict__ out, long long total)
{
    long long i = (long long)blockIdx.x * 256 + threadIdx.x;
    if (i >= total) return;
    float v = hp[i];
    out[i] = (v > 0.f ? v : expf(v) - 1.0f) + h[i];
}

__global__ void gin_build_kernel(const float* __restrict__ avg, const float* __restrict__ mx,
                                 float* __restrict__ gin, int Fdim, long long total)
{
    long long i = (long long)blockIdx.x * 256 + threadIdx.x;
    if (i >= total) return;
    long long row = i / (2 * Fdim);
    int c = (int)(i % (2 * Fdim));
    gin[i] = (c < Fdim) ? avg[row * Fdim + c] : mx[row * Fdim + (c - Fdim)];
}

__global__ void fuse_gate_kernel(const float* __restrict__ g, const float* __restrict__ avg,
                                 const float* __restrict__ mx, float* __restrict__ out,
                                 long long total)
{
    long long i = (long long)blockIdx.x * 256 + threadIdx.x;
    if (i >= total) return;
    float gv = g[i];
    out[i] = gv * avg[i] + (1.0f - gv) * mx[i];
}

// ---------------------------------------------------------------------------
// Host orchestration
// ---------------------------------------------------------------------------
extern "C" void kernel_launch(void* const* d_in, const int* in_sizes, int n_in,
                              void* d_out, int out_size, void* d_ws, size_t ws_size,
                              hipStream_t stream)
{
    const int Bn = 32, Nn = 512, Fd = 512, Hh = 4, Pp = 12;
    const int M  = Bn * Nn;                        // 16384
    const long long SZ  = (long long)Bn * Nn * Fd; // per-tensor floats (== B*N*N)
    const long long SZb = (long long)Nn * Fd;      // per-batch stride
    const int FF = Fd * Fd;

    const float* x_real = (const float*)d_in[0];
    const float* x_imag = (const float*)d_in[1];
    const float* adj    = (const float*)d_in[2];
    const float* W_q    = (const float*)d_in[3];
    const float* W_k    = (const float*)d_in[4];
    const float* W_v    = (const float*)d_in[5];
    const float* att_w  = (const float*)d_in[6];
    const float* att_b  = (const float*)d_in[7];
    const float* gat_W  = (const float*)d_in[8];
    const float* gat_a  = (const float*)d_in[9];
    const float* node_w = (const float*)d_in[10];
    const float* node_b = (const float*)d_in[11];
    const float* gln_gh = (const float*)d_in[12];
    const float* gln_bh = (const float*)d_in[13];
    const float* norm_g = (const float*)d_in[14];
    const float* norm_b = (const float*)d_in[15];
    const float* g1_w   = (const float*)d_in[16];
    const float* g1_b   = (const float*)d_in[17];
    const float* gl_g   = (const float*)d_in[18];
    const float* gl_b   = (const float*)d_in[19];
    const float* g2_w   = (const float*)d_in[20];
    const float* g2_b   = (const float*)d_in[21];
    const float* enc_w  = (const float*)d_in[22];
    const float* enc_b  = (const float*)d_in[23];
    const float* dec_w  = (const float*)d_in[24];
    const float* dec_b  = (const float*)d_in[25];
    float* out = (float*)d_out;

    float* W = (float*)d_ws;
    float* S0 = W + 0 * SZ;  float* S1 = W + 1 * SZ;
    float* S2 = W + 2 * SZ;  float* S3 = W + 3 * SZ;
    float* S4 = W + 4 * SZ;  float* S5 = W + 5 * SZ;
    float* S6 = W + 6 * SZ;  float* S7 = W + 7 * SZ;
    float* e1b = W + 8 * SZ;
    float* e2b = e1b + M;

    auto gemm = [&](const float* A, const float* Bp, float* Cp, int M_, int Nc_, int K_,
                    int lda, int ldb, int ldc, int batch,
                    long long sA, long long sB, long long sC,
                    float alpha, int accum, int transB, const float* bias, int mode) {
        dim3 grid((Nc_ + BN - 1) / BN, (M_ + BM - 1) / BM, batch);
        if (transB)
            gemm_bf16_wmma<1, 0><<<grid, 256, 0, stream>>>(A, Bp, Cp, M_, Nc_, K_, lda, ldb, ldc,
                                                           sA, sB, sC, alpha, accum, bias, mode);
        else if (Nc_ % BN != 0)
            gemm_bf16_wmma<0, 1><<<grid, 256, 0, stream>>>(A, Bp, Cp, M_, Nc_, K_, lda, ldb, ldc,
                                                           sA, sB, sC, alpha, accum, bias, mode);
        else
            gemm_bf16_wmma<0, 0><<<grid, 256, 0, stream>>>(A, Bp, Cp, M_, Nc_, K_, lda, ldb, ldc,
                                                           sA, sB, sC, alpha, accum, bias, mode);
    };
    auto ew_blocks = [](long long n) { return (unsigned)((n + 255) / 256); };
    const dim3 lnGrid(M / 8);

    // ---- complex projections: Qr,Qi,Kr,Ki,Vr,Vi ----
    gemm(x_real, W_q,      S0, M, Fd, Fd, Fd, Fd, Fd, 1, 0, 0, 0,  1.f, 0, 0, nullptr, EP_NONE);
    gemm(x_imag, W_q + FF, S0, M, Fd, Fd, Fd, Fd, Fd, 1, 0, 0, 0, -1.f, 1, 0, nullptr, EP_NONE);
    gemm(x_imag, W_q,      S1, M, Fd, Fd, Fd, Fd, Fd, 1, 0, 0, 0,  1.f, 0, 0, nullptr, EP_NONE);
    gemm(x_real, W_q + FF, S1, M, Fd, Fd, Fd, Fd, Fd, 1, 0, 0, 0,  1.f, 1, 0, nullptr, EP_NONE);
    gemm(x_real, W_k,      S2, M, Fd, Fd, Fd, Fd, Fd, 1, 0, 0, 0,  1.f, 0, 0, nullptr, EP_NONE);
    gemm(x_imag, W_k + FF, S2, M, Fd, Fd, Fd, Fd, Fd, 1, 0, 0, 0, -1.f, 1, 0, nullptr, EP_NONE);
    gemm(x_imag, W_k,      S3, M, Fd, Fd, Fd, Fd, Fd, 1, 0, 0, 0,  1.f, 0, 0, nullptr, EP_NONE);
    gemm(x_real, W_k + FF, S3, M, Fd, Fd, Fd, Fd, Fd, 1, 0, 0, 0,  1.f, 1, 0, nullptr, EP_NONE);
    gemm(x_real, W_v,      S4, M, Fd, Fd, Fd, Fd, Fd, 1, 0, 0, 0,  1.f, 0, 0, nullptr, EP_NONE);
    gemm(x_imag, W_v + FF, S4, M, Fd, Fd, Fd, Fd, Fd, 1, 0, 0, 0, -1.f, 1, 0, nullptr, EP_NONE);
    gemm(x_imag, W_v,      S5, M, Fd, Fd, Fd, Fd, Fd, 1, 0, 0, 0,  1.f, 0, 0, nullptr, EP_NONE);
    gemm(x_real, W_v + FF, S5, M, Fd, Fd, Fd, Fd, Fd, 1, 0, 0, 0,  1.f, 1, 0, nullptr, EP_NONE);

    // ---- complex attention scores (batched, Q·K^T), fused causal mask + lrelu ----
    const float scl = 1.0f / sqrtf((float)Nn);
    gemm(S0, S2, S6, Nn, Nn, Fd, Fd, Fd, Nn, Bn, SZb, SZb, SZb,  scl, 0, 1, nullptr, EP_NONE);
    gemm(S1, S3, S6, Nn, Nn, Fd, Fd, Fd, Nn, Bn, SZb, SZb, SZb, -scl, 1, 1, nullptr, EP_CAUSAL);
    gemm(S0, S3, S7, Nn, Nn, Fd, Fd, Fd, Nn, Bn, SZb, SZb, SZb,  scl, 0, 1, nullptr, EP_NONE);
    gemm(S1, S2, S7, Nn, Nn, Fd, Fd, Fd, Nn, Bn, SZb, SZb, SZb,  scl, 1, 1, nullptr, EP_CAUSAL);

    // ---- apply attention + att_map + encoder combine ----
    gemm(S6, S4, S0, Nn, Fd, Nn, Nn, Fd, Fd, Bn, SZb, SZb, SZb, 1.f, 0, 0, nullptr, EP_NONE);
    gemm(S0, att_w, S1, M, Fd, Fd, Fd, Fd, Fd, 1, 0, 0, 0, 1.f, 0, 0, att_b, EP_NONE);
    gemm(S7, S5, S2, Nn, Fd, Nn, Nn, Fd, Fd, Bn, SZb, SZb, SZb, 1.f, 0, 0, nullptr, EP_NONE);
    gemm(S2, att_w, S3, M, Fd, Fd, Fd, Fd, Fd, 1, 0, 0, 0, 1.f, 0, 0, att_b, EP_NONE);
    enc_combine_kernel<<<ew_blocks(SZ), 256, 0, stream>>>(S1, S3, enc_w, enc_b, S4, SZ);

    // ---- GAT heads, streaming avg/max reduction ----
    const long long BNN = (long long)Bn * Nn * Nn;
    for (int h = 0; h < Hh; ++h) {
        gat_input_kernel<<<ew_blocks(SZ), 256, 0, stream>>>(S4, node_w + h * Nn, node_b + h * Nn,
                                                            S0, Nn, Fd, SZ);
        gemm(S0, gat_W + (long long)h * FF, S1, M, Fd, Fd, Fd, Fd, Fd, 1, 0, 0, 0,
             1.f, 0, 0, nullptr, EP_NONE);
        ln_kernel<<<lnGrid, 256, 0, stream>>>(S1, gln_gh + h * Fd, gln_bh + h * Fd,
                                              S2, nullptr, nullptr, Fd, 0);
        gemv2_kernel<<<lnGrid, 256, 0, stream>>>(S2, gat_a + (long long)h * 2 * Fd,
                                                 gat_a + (long long)h * 2 * Fd + Fd,
                                                 e1b, e2b, Fd);
        att_build_kernel<<<ew_blocks(BNN), 256, 0, stream>>>(e1b, e2b, adj, S0, Nn, BNN);
        gemm(S0, S2, S1, Nn, Fd, Nn, Nn, Fd, Fd, Bn, SZb, SZb, SZb, 1.f, 0, 0, nullptr, EP_NONE);
        elu_res_kernel<<<ew_blocks(SZ), 256, 0, stream>>>(S1, S2, S3, SZ);
        ln_kernel<<<lnGrid, 256, 0, stream>>>(S3, norm_g, norm_b, nullptr, S5, S6, Fd,
                                              (h == 0) ? 2 : 3);
    }

    // ---- gate MLP + fuse + decoder ----
    gin_build_kernel<<<ew_blocks(2 * SZ), 256, 0, stream>>>(S5, S6, S0, Fd, 2 * SZ); // gin in S0..S1
    gemm(S0, g1_w, S2, M, Fd, 2 * Fd, 2 * Fd, Fd, Fd, 1, 0, 0, 0, 1.f, 0, 0, g1_b, EP_NONE);
    ln_kernel<<<lnGrid, 256, 0, stream>>>(S2, gl_g, gl_b, S3, nullptr, nullptr, Fd, 1); // LN+GELU
    gemm(S3, g2_w, S4, M, Fd, Fd, Fd, Fd, Fd, 1, 0, 0, 0, 1.f, 0, 0, g2_b, EP_SIGM);
    fuse_gate_kernel<<<ew_blocks(SZ), 256, 0, stream>>>(S4, S5, S6, S2, SZ);
    gemm(S2, dec_w, out, M, Pp, Fd, Fd, Pp, Pp, 1, 0, 0, 0, 1.f, 0, 0, dec_b, EP_NONE);
}
